// LSTMClassifier_59261958750824
// MI455X (gfx1250) — matrix operand order repaired and checked by reading, not compiled, not run on hardware
//
#include <hip/hip_runtime.h>
#include <hip/hip_bf16.h>
#include <math.h>

typedef __attribute__((ext_vector_type(16))) __bf16 v16bf;
typedef __attribute__((ext_vector_type(8)))  __bf16 v8bf;
typedef __attribute__((ext_vector_type(8)))  float  v8f;

#define B_   128
#define T_   512
#define F_   3
#define U_   512
#define C_   5
#define G4   2048          // 4*U
#define NWG  32            // persistent workgroups for the recurrence
#define UPW  16            // units per WG
#define WCOLS 64           // 4 gates * UPW columns per WG
#define WLDK 520           // padded K stride in LDS (bank-spread)
#define GB_COLS 64         // columns per block in the big GEMM

__device__ __forceinline__ float sigmoidf_(float x){ return 1.0f/(1.0f+__expf(-x)); }

__device__ __forceinline__ v16bf cat8(v8bf lo, v8bf hi){
  return __builtin_shufflevector(lo, hi, 0,1,2,3,4,5,6,7,8,9,10,11,12,13,14,15);
}

// A fragment (16x32 bf16, M x K): lane&15 = row, lane>>4 selects K-half.
// Per ISA: half0 lanes hold K {0..7,16..23}, half1 lanes K {8..15,24..31}.
__device__ __forceinline__ v16bf load_a_bf16(const __bf16* base, int row, int k0, int lane){
  const __bf16* p = base + (size_t)row*U_ + k0 + ((lane>>4)<<3);
  v8bf r0 = *(const v8bf*)(p);        // K = k0+half*8 .. +7
  v8bf r1 = *(const v8bf*)(p + 16);   // K = k0+16+half*8 .. +7
  return cat8(r0, r1);
}
// Same fragment but sourced from fp32 rows (convert to bf16 on the fly).
__device__ __forceinline__ v16bf load_a_f32(const float* base, int ldk, int row, int k0, int lane){
  const float* p = base + (size_t)row*ldk + k0 + ((lane>>4)<<3);
  v16bf a;
#pragma unroll
  for(int e=0;e<8;++e)  a[e]   = (__bf16)p[e];
#pragma unroll
  for(int e=0;e<8;++e)  a[8+e] = (__bf16)p[16+e];
  return a;
}
// B fragment (32x16 bf16, K x N) from LDS tile stored [col][K] (stride WLDK):
// lane&15 = col, lanes<16 hold K {k0..k0+7, k0+16..k0+23}, lanes>=16 hold K {k0+8..k0+15, k0+24..k0+31}.
__device__ __forceinline__ v16bf load_b_lds(const __bf16* w, int col0, int k0, int lane){
  const __bf16* p = w + (size_t)(col0 + (lane&15))*WLDK + k0 + ((lane>>4)<<3);
  v8bf r0 = *(const v8bf*)(p);
  v8bf r1 = *(const v8bf*)(p + 16);
  return cat8(r0, r1);
}

// Device-wide sense/generation barrier for the persistent recurrence.
__device__ __forceinline__ void grid_barrier(int* cnt, int* gen, int nwg){
  __syncthreads();
  if(threadIdx.x==0){
    __threadfence();
    int g = __hip_atomic_load(gen, __ATOMIC_ACQUIRE, __HIP_MEMORY_SCOPE_AGENT);
    int a = __hip_atomic_fetch_add(cnt, 1, __ATOMIC_ACQ_REL, __HIP_MEMORY_SCOPE_AGENT);
    if(a == nwg-1){
      __hip_atomic_store(cnt, 0, __ATOMIC_RELAXED, __HIP_MEMORY_SCOPE_AGENT);
      __hip_atomic_fetch_add(gen, 1, __ATOMIC_ACQ_REL, __HIP_MEMORY_SCOPE_AGENT);
    } else {
      while(__hip_atomic_load(gen, __ATOMIC_ACQUIRE, __HIP_MEMORY_SCOPE_AGENT) == g)
        __builtin_amdgcn_s_sleep(1);
    }
  }
  __syncthreads();
}

// ---------------- zx = x @ Wk0 + b0  (K = 3, streaming) ----------------
__global__ __launch_bounds__(256) void zx0_kernel(const float* __restrict__ x,
                                                  const float* __restrict__ Wk0,
                                                  const float* __restrict__ b0,
                                                  float* __restrict__ zx){
  size_t i = (size_t)blockIdx.x*256 + threadIdx.x;
  int    col = (int)(i & (G4-1));
  size_t row = i >> 11;
  float acc = b0[col];
#pragma unroll
  for(int f=0; f<F_; ++f) acc += x[row*F_+f] * Wk0[f*G4+col];
  zx[i] = acc;
}

__global__ void init_ctrl_kernel(int* ctrl){ if(threadIdx.x < 2) ctrl[threadIdx.x] = 0; }

// ---------------- persistent masked LSTM recurrence (WMMA bf16) ----------------
// Each of NWG=32 blocks owns UPW=16 units: z-tile [128 x 64] per step.
// 8 waves: wave w computes C rows 16w..16w+15 for all 4 gate tiles.
// zx[t] is software-pipelined one step ahead in registers (zv) so the
// HBM-resident zx stream never sits on the per-step critical path.
__global__ __launch_bounds__(256) void lstm_rec_kernel(
    const float* __restrict__ zx,            // [B*T, 2048] precomputed x-projection (+bias)
    const float* __restrict__ Wr,            // [512, 2048] fp32 recurrent weights
    const unsigned char* __restrict__ masks, // [B, T]
    float* __restrict__ hs,                  // [B, T, 512] output sequence
    __bf16* __restrict__ hbf,                // [2][128*512] double-buffered h (bf16)
    int* __restrict__ ctrl)                  // {cnt, gen}
{
  __shared__ __bf16 wlds[WCOLS*WLDK];
  const int tid = threadIdx.x, wg = blockIdx.x;
  const int wav = tid>>5, lane = tid&31;
  const int u0 = wg*UPW;

  // Stage our Wr slice fp32 -> bf16 into LDS, [col][K] padded (once for all 512 steps).
  for(int idx=tid; idx<WCOLS*U_; idx+=256){
    int col = idx >> 9, k = idx & (U_-1);
    int gate = col >> 4, u = u0 + (col & 15);
    wlds[(size_t)col*WLDK + k] = (__bf16)Wr[(size_t)k*G4 + gate*U_ + u];
  }
  // Zero h_{-1} (buffer 0): each WG zeroes its unit columns for all batches.
  for(int idx=tid; idx<B_*UPW; idx+=256){
    int b = idx >> 4, j = idx & 15;
    hbf[(size_t)b*U_ + u0 + j] = (__bf16)0.0f;
  }
  __syncthreads();
  grid_barrier(ctrl, ctrl+1, NWG);

  float c8[8], h8[8];
#pragma unroll
  for(int g=0; g<8; ++g){ c8[g]=0.f; h8[g]=0.f; }

  const int half  = lane>>4;
  const int arow  = 16*wav + (lane&15);   // A-matrix row (batch) for this lane
  const int ccol  = lane & 15;            // C-matrix column (unit offset)
  const int crow0 = 16*wav + 8*half;      // C-matrix row base (add g)

  // Preload zx for t=0 (pays HBM latency exactly once).
  float zv[4][8];
#pragma unroll
  for(int gt=0; gt<4; ++gt)
#pragma unroll
    for(int g=0; g<8; ++g){
      int b = crow0 + g;
      zv[gt][g] = zx[((size_t)b*T_)*G4 + gt*U_ + u0 + ccol];
    }

  for(int t=0; t<T_; ++t){
    const __bf16* hcur = hbf + (size_t)(t&1)*B_*U_;
    __bf16*       hnxt = hbf + (size_t)((t+1)&1)*B_*U_;

    // Seed accumulators from registers (zx[t] was fetched last step).
    v8f acc[4];
#pragma unroll
    for(int gt=0; gt<4; ++gt)
#pragma unroll
      for(int g=0; g<8; ++g) acc[gt][g] = zv[gt][g];

#pragma unroll 4
    for(int kc=0; kc<U_/32; ++kc){        // h_{t-1} @ Wr_slice
      v16bf a = load_a_bf16(hcur, arow, kc*32, lane);
#pragma unroll
      for(int gt=0; gt<4; ++gt){
        v16bf b = load_b_lds(wlds, gt*16, kc*32, lane);
        acc[gt] = __builtin_amdgcn_wmma_f32_16x16x32_bf16(
            false, a, false, b, (short)0, acc[gt], false, false);
      }
    }

    // Issue next step's zx loads now: they complete during the gate math,
    // the global barrier, and the next step's K loop.
    if(t+1 < T_){
#pragma unroll
      for(int gt=0; gt<4; ++gt)
#pragma unroll
        for(int g=0; g<8; ++g){
          int b = crow0 + g;
          zv[gt][g] = zx[((size_t)b*T_ + (t+1))*G4 + gt*U_ + u0 + ccol];
        }
    }

#pragma unroll
    for(int g=0; g<8; ++g){
      int b = crow0 + g;
      float iv = sigmoidf_(acc[0][g]);
      float fv = sigmoidf_(acc[1][g]);
      float gv = tanhf(acc[2][g]);
      float ov = sigmoidf_(acc[3][g]);
      float cn = fv*c8[g] + iv*gv;
      float hn = ov*tanhf(cn);
      bool m = masks[(size_t)b*T_ + t] != 0;
      c8[g] = m ? cn : c8[g];
      h8[g] = m ? hn : h8[g];
      hs[((size_t)b*T_ + t)*U_ + u0 + ccol] = h8[g];
      hnxt[(size_t)b*U_ + u0 + ccol] = (__bf16)h8[g];
    }
    grid_barrier(ctrl, ctrl+1, NWG);
  }
}

// ---------------- layer norm in place over U=512 ----------------
__global__ __launch_bounds__(256) void ln_kernel(float* __restrict__ h,
                                                 const float* __restrict__ gamma,
                                                 const float* __restrict__ beta){
  __shared__ float red[256];
  float* p = h + (size_t)blockIdx.x*U_;
  int tid = threadIdx.x;
  float v0 = p[tid], v1 = p[tid+256];
  red[tid] = v0+v1; __syncthreads();
  for(int s=128;s>0;s>>=1){ if(tid<s) red[tid]+=red[tid+s]; __syncthreads(); }
  float mu = red[0]*(1.0f/U_); __syncthreads();
  float d0 = v0-mu, d1 = v1-mu;
  red[tid] = d0*d0+d1*d1; __syncthreads();
  for(int s=128;s>0;s>>=1){ if(tid<s) red[tid]+=red[tid+s]; __syncthreads(); }
  float rs = rsqrtf(red[0]*(1.0f/U_) + 0.001f);
  p[tid]     = d0*rs*gamma[tid]     + beta[tid];
  p[tid+256] = d1*rs*gamma[tid+256] + beta[tid+256];
}

// ---------------- zx1 = LN(h0) @ Wk1 + b1  (65536x2048x512 WMMA GEMM) ----------------
__global__ __launch_bounds__(256) void zx1_gemm_kernel(const float* __restrict__ A,   // [65536,512]
                                                       const float* __restrict__ Wk,  // [512,2048]
                                                       const float* __restrict__ bias,// [2048]
                                                       float* __restrict__ out){      // [65536,2048]
  __shared__ __bf16 wlds[GB_COLS*WLDK];
  const int tid = threadIdx.x, wav = tid>>5, lane = tid&31;
  const int gc0 = blockIdx.x*GB_COLS;
  const int m0base = blockIdx.y * ((B_*T_)/64);   // 1024 rows per block
  for(int idx=tid; idx<GB_COLS*U_; idx+=256){
    int col = idx>>9, k = idx & (U_-1);
    wlds[(size_t)col*WLDK + k] = (__bf16)Wk[(size_t)k*G4 + gc0 + col];
  }
  __syncthreads();
  const int half = lane>>4;
  const int arow_off = 16*wav + (lane&15);
  const int ccol = lane & 15;
  const int crow_off = 16*wav + 8*half;
  for(int mt=0; mt<8; ++mt){
    int m0 = m0base + mt*128;
    v8f acc[4];
#pragma unroll
    for(int nt=0; nt<4; ++nt){
      float bv = bias[gc0 + nt*16 + ccol];
#pragma unroll
      for(int g=0; g<8; ++g) acc[nt][g] = bv;
    }
#pragma unroll 4
    for(int kc=0; kc<U_/32; ++kc){
      v16bf a = load_a_f32(A, U_, m0 + arow_off, kc*32, lane);
#pragma unroll
      for(int nt=0; nt<4; ++nt){
        v16bf b = load_b_lds(wlds, nt*16, kc*32, lane);
        acc[nt] = __builtin_amdgcn_wmma_f32_16x16x32_bf16(
            false, a, false, b, (short)0, acc[nt], false, false);
      }
    }
#pragma unroll
    for(int nt=0; nt<4; ++nt)
#pragma unroll
      for(int g=0; g<8; ++g)
        out[(size_t)(m0+crow_off+g)*G4 + gc0 + nt*16 + ccol] = acc[nt][g];
  }
}

// ---------------- fused LN + FC(512->5) + softmax ----------------
__global__ __launch_bounds__(256) void fc_softmax_kernel(const float* __restrict__ h,
    const float* __restrict__ gamma, const float* __restrict__ beta,
    const float* __restrict__ Wfc, const float* __restrict__ bfc,
    float* __restrict__ out){
  __shared__ float red[256];
  __shared__ float logits[C_];
  const float* p = h + (size_t)blockIdx.x*U_;
  int tid = threadIdx.x;
  float v0 = p[tid], v1 = p[tid+256];
  red[tid] = v0+v1; __syncthreads();
  for(int s=128;s>0;s>>=1){ if(tid<s) red[tid]+=red[tid+s]; __syncthreads(); }
  float mu = red[0]*(1.0f/U_); __syncthreads();
  float d0 = v0-mu, d1 = v1-mu;
  red[tid] = d0*d0+d1*d1; __syncthreads();
  for(int s=128;s>0;s>>=1){ if(tid<s) red[tid]+=red[tid+s]; __syncthreads(); }
  float rs = rsqrtf(red[0]*(1.0f/U_) + 0.001f); __syncthreads();
  float y0 = d0*rs*gamma[tid]     + beta[tid];
  float y1 = d1*rs*gamma[tid+256] + beta[tid+256];
  for(int c=0;c<C_;++c){
    red[tid] = y0*Wfc[tid*C_+c] + y1*Wfc[(tid+256)*C_+c];
    __syncthreads();
    for(int s=128;s>0;s>>=1){ if(tid<s) red[tid]+=red[tid+s]; __syncthreads(); }
    if(tid==0) logits[c] = red[0] + bfc[c];
    __syncthreads();
  }
  if(tid==0){
    float mx = logits[0];
    for(int c=1;c<C_;++c) mx = fmaxf(mx, logits[c]);
    float s = 0.f, e[C_];
    for(int c=0;c<C_;++c){ e[c] = __expf(logits[c]-mx); s += e[c]; }
    for(int c=0;c<C_;++c) out[(size_t)blockIdx.x*C_ + c] = e[c]/s;
  }
}

extern "C" void kernel_launch(void* const* d_in, const int* in_sizes, int n_in,
                              void* d_out, int out_size, void* d_ws, size_t ws_size,
                              hipStream_t stream) {
  const float* x     = (const float*)d_in[0];
  const unsigned char* masks = (const unsigned char*)d_in[1];
  const float* Wk0   = (const float*)d_in[2];
  const float* Wr0   = (const float*)d_in[3];
  const float* b0    = (const float*)d_in[4];
  const float* Wk1   = (const float*)d_in[5];
  const float* Wr1   = (const float*)d_in[6];
  const float* b1    = (const float*)d_in[7];
  const float* gamma = (const float*)d_in[8];
  const float* beta  = (const float*)d_in[9];
  const float* Wfc   = (const float*)d_in[10];
  const float* bfc   = (const float*)d_in[11];
  float* out = (float*)d_out;
  (void)in_sizes; (void)n_in; (void)out_size; (void)ws_size;

  const size_t ZX_BYTES  = (size_t)B_*T_*G4*sizeof(float);   // 512 MiB
  const size_t HS_BYTES  = (size_t)B_*T_*U_*sizeof(float);   // 128 MiB
  const size_t HBF_BYTES = (size_t)2*B_*U_*2;                // 256 KiB
  char* w = (char*)d_ws;
  float*  zx   = (float*)w;
  float*  hs   = (float*)(w + ZX_BYTES);
  __bf16* hbf  = (__bf16*)(w + ZX_BYTES + HS_BYTES);
  int*    ctrl = (int*)(w + ZX_BYTES + HS_BYTES + HBF_BYTES);

  const int rows = B_*T_;                      // 65536
  // 1) zx = x @ Wk0 + b0
  zx0_kernel<<<(unsigned)((size_t)rows*G4/256), 256, 0, stream>>>(x, Wk0, b0, zx);
  // 2) layer-0 recurrence
  init_ctrl_kernel<<<1, 64, 0, stream>>>(ctrl);
  lstm_rec_kernel<<<NWG, 256, 0, stream>>>(zx, Wr0, masks, hs, hbf, ctrl);
  // 3) LN in place
  ln_kernel<<<rows, 256, 0, stream>>>(hs, gamma, beta);
  // 4) zx = LN(h0) @ Wk1 + b1 (reuse zx buffer)
  zx1_gemm_kernel<<<dim3(G4/GB_COLS, 64), 256, 0, stream>>>(hs, Wk1, b1, zx);
  // 5) layer-1 recurrence (overwrite hs)
  init_ctrl_kernel<<<1, 64, 0, stream>>>(ctrl);
  lstm_rec_kernel<<<NWG, 256, 0, stream>>>(zx, Wr1, masks, hs, hbf, ctrl);
  // 6) fused LN + FC + softmax
  fc_softmax_kernel<<<rows, 256, 0, stream>>>(hs, gamma, beta, Wfc, bfc, out);
}
